// PostNetwork_18442589570221
// MI455X (gfx1250) — matrix operand order repaired and checked by reading, not compiled, not run on hardware
//
#include <hip/hip_runtime.h>
#include <hip/hip_bf16.h>

typedef __bf16 bf16;
typedef __attribute__((ext_vector_type(16))) __bf16 v16bf;
typedef __attribute__((ext_vector_type(8)))  float  v8f;
typedef __attribute__((ext_vector_type(4)))  unsigned int v4u;

union Frag16 { v16bf v; bf16 h[16]; v4u u[2]; };

#define DIMK 128
#define WAVES_PER_BLOCK 8
#define EDGES_PER_WAVE 32                       // two 16-edge B tiles per wave
#define EDGES_PER_BLOCK (WAVES_PER_BLOCK * EDGES_PER_WAVE)   // 256

// ---- LDS layout (bytes): weights only, no h staging ----
#define LDS_W1T   0                              // W1^T [128 out][256 in] bf16 = 65536
#define LDS_W2T   65536                          // W2^T [128 out][128 in] bf16 = 32768
#define LDS_W3    (65536 + 32768)                // [128] bf16 = 256
#define LDS_B1    (LDS_W3 + 256)                 // [128] f32 = 512
#define LDS_B2    (LDS_B1 + 512)                 // [128] f32 = 512
#define LDS_B3    (LDS_B2 + 512)                 // f32 (+pad)
#define LDS_TOTAL (LDS_B3 + 128)                 // ~99.7 KB -> 3 workgroups / WGP

static_assert(LDS_TOTAL <= 108 * 1024, "3 workgroups per WGP (320KB LDS)");

extern __shared__ char smem[];

__device__ inline v8f wmma_bf16(const Frag16& a, const Frag16& b, v8f c) {
    return __builtin_amdgcn_wmma_f32_16x16x32_bf16(
        false, a.v, false, b.v, (short)0, c, false, false);
}

// 16 consecutive fp32 -> B-fragment elements 0..15 (K ascending)
__device__ inline void cvt16(const float* __restrict__ p, Frag16& f) {
    const float4* q = (const float4*)p;
    float4 a = q[0], b = q[1], c = q[2], d = q[3];
    f.h[0]  = (bf16)a.x; f.h[1]  = (bf16)a.y; f.h[2]  = (bf16)a.z; f.h[3]  = (bf16)a.w;
    f.h[4]  = (bf16)b.x; f.h[5]  = (bf16)b.y; f.h[6]  = (bf16)b.z; f.h[7]  = (bf16)b.w;
    f.h[8]  = (bf16)c.x; f.h[9]  = (bf16)c.y; f.h[10] = (bf16)c.z; f.h[11] = (bf16)c.w;
    f.h[12] = (bf16)d.x; f.h[13] = (bf16)d.y; f.h[14] = (bf16)d.z; f.h[15] = (bf16)d.w;
}

// relu + pack 8 f32 accumulator lanes (features ascending) into 8 bf16 (4 dwords)
__device__ inline v4u relu_pack8(v8f a) {
    union { bf16 h[8]; v4u u; } t;
#pragma unroll
    for (int j = 0; j < 8; ++j) {
        float v = a[j];
        t.h[j] = (bf16)(v > 0.f ? v : 0.f);
    }
    return t.u;
}

__device__ inline v4u shfl_xor16(v4u x) {
    v4u r;
    r.x = (unsigned)__shfl_xor((int)x.x, 16, 32);
    r.y = (unsigned)__shfl_xor((int)x.y, 16, 32);
    r.z = (unsigned)__shfl_xor((int)x.z, 16, 32);
    r.w = (unsigned)__shfl_xor((int)x.w, 16, 32);
    return r;
}

// Build next-layer B fragment (K = 32*kk + 16*khalf + 0..15 for this lane's edge
// column) from this lane's packed D runs + partner lane's (lane ^ 16) runs.
__device__ inline Frag16 mkB(const v4u* own, int kk, int khalf) {
    v4u pA = shfl_xor16(own[2 * kk]);       // partner's features [16*(2kk)+8*kh', +8)
    v4u pB = shfl_xor16(own[2 * kk + 1]);
    Frag16 b;
    b.u[0] = khalf ? pB : own[2 * kk];      // K lo half
    b.u[1] = khalf ? own[2 * kk + 1] : pA;  // K hi half
    return b;
}

__global__ __launch_bounds__(256)
void edge_mlp_wmma(const float* __restrict__ x1,
                   const float* __restrict__ x2,
                   const long long* __restrict__ ei,
                   const float* __restrict__ W1, const float* __restrict__ b1,
                   const float* __restrict__ W2, const float* __restrict__ b2,
                   const float* __restrict__ W3, const float* __restrict__ b3,
                   float* __restrict__ out, long long nE)
{
    bf16*  W1t = (bf16*)(smem + LDS_W1T);
    bf16*  W2t = (bf16*)(smem + LDS_W2T);
    bf16*  w3l = (bf16*)(smem + LDS_W3);
    float* b1l = (float*)(smem + LDS_B1);
    float* b2l = (float*)(smem + LDS_B2);
    float* b3l = (float*)(smem + LDS_B3);

    const int t = threadIdx.x;

    // ---- Stage weights (transposed, bf16) into LDS ----
    for (int i = t; i < 256 * DIMK; i += 256) {      // W1 [256][128] row-major
        int k = i >> 7, n = i & 127;
        W1t[n * 256 + k] = (bf16)W1[i];
    }
    for (int i = t; i < DIMK * DIMK; i += 256) {     // W2 [128][128] row-major
        int k = i >> 7, n = i & 127;
        W2t[n * DIMK + k] = (bf16)W2[i];
    }
    if (t < DIMK) { w3l[t] = (bf16)W3[t]; b1l[t] = b1[t]; b2l[t] = b2[t]; }
    if (t == 0)   { b3l[0] = b3[0]; }
    __syncthreads();

    const int lane  = t & 31;
    const int wave  = t >> 5;
    const int row   = lane & 15;   // A: M row (out feature); B/D: N column (edge)
    const int khalf = lane >> 4;

    const long long edge0 = (long long)blockIdx.x * EDGES_PER_BLOCK
                          + (long long)wave * EDGES_PER_WAVE;
    long long e0 = edge0 + row;
    long long e1 = edge0 + 16 + row;
    long long e0c = (e0 < nE) ? e0 : (nE - 1);       // clamp: EXEC stays all-1s
    long long e1c = (e1 < nE) ? e1 : (nE - 1);
    const float* xs0 = x1 + ei[e0c] * DIMK;
    const float* xt0 = x2 + ei[nE + e0c] * DIMK;
    const float* xs1 = x1 + ei[e1c] * DIMK;
    const float* xt1 = x2 + ei[nE + e1c] * DIMK;

    // ================= Layer 1: D = W1^T(128x256) x pair(256x32 edges) + b1 ======
    v8f acc0[8], acc1[8];
#pragma unroll
    for (int mm = 0; mm < 8; ++mm) {
        v8f bv = *(const v8f*)(b1l + mm * 16 + khalf * 8);  // features 16mm+8kh+0..7
        acc0[mm] = bv;
        acc1[mm] = bv;
    }
#pragma unroll
    for (int kk = 0; kk < 8; ++kk) {
        // Batch-load the 8 weight A-fragments for this K step (ds_load_b128 x16)
        Frag16 A[8];
#pragma unroll
        for (int mm = 0; mm < 8; ++mm) {
            const bf16* wp = W1t + (mm * 16 + row) * 256 + kk * 32 + khalf * 16;
            A[mm].u[0] = *(const v4u*)wp;          // K run {kk*32+16kh+0..7}
            A[mm].u[1] = *(const v4u*)(wp + 8);    // K run {kk*32+16kh+8..15}
        }
        // Gather edge B-fragments: 16 consecutive feats of pair row
        const int coff = (kk & 3) * 32 + khalf * 16;
        Frag16 B0, B1;
        cvt16(((kk < 4) ? xs0 : xt0) + coff, B0);
        cvt16(((kk < 4) ? xs1 : xt1) + coff, B1);
#pragma unroll
        for (int mm = 0; mm < 8; ++mm) {
            acc0[mm] = wmma_bf16(A[mm], B0, acc0[mm]);
            acc1[mm] = wmma_bf16(A[mm], B1, acc1[mm]);
        }
    }

    // ---- transition 1: relu + bf16 pack; h never touches LDS ----
    v4u own0[8], own1[8];
#pragma unroll
    for (int mm = 0; mm < 8; ++mm) { own0[mm] = relu_pack8(acc0[mm]); }
#pragma unroll
    for (int mm = 0; mm < 8; ++mm) { own1[mm] = relu_pack8(acc1[mm]); }

    // ================= Layer 2: D = W2^T(128x128) x h(128x32) + b2 ===============
    v8f c20[8], c21[8];
#pragma unroll
    for (int mm = 0; mm < 8; ++mm) {
        v8f bv = *(const v8f*)(b2l + mm * 16 + khalf * 8);
        c20[mm] = bv;
        c21[mm] = bv;
    }
#pragma unroll
    for (int kk = 0; kk < 4; ++kk) {
        Frag16 A[8];
#pragma unroll
        for (int mm = 0; mm < 8; ++mm) {
            const bf16* wp = W2t + (mm * 16 + row) * DIMK + kk * 32 + khalf * 16;
            A[mm].u[0] = *(const v4u*)wp;
            A[mm].u[1] = *(const v4u*)(wp + 8);
        }
        Frag16 B0 = mkB(own0, kk, khalf);
        Frag16 B1 = mkB(own1, kk, khalf);
#pragma unroll
        for (int mm = 0; mm < 8; ++mm) {
            c20[mm] = wmma_bf16(A[mm], B0, c20[mm]);
            c21[mm] = wmma_bf16(A[mm], B1, c21[mm]);
        }
    }

    // ---- transition 2 ----
    v4u o20[8], o21[8];
#pragma unroll
    for (int mm = 0; mm < 8; ++mm) { o20[mm] = relu_pack8(c20[mm]); }
#pragma unroll
    for (int mm = 0; mm < 8; ++mm) { o21[mm] = relu_pack8(c21[mm]); }

    // ================= Layer 3: D = w3^T(1x128) x h(128x32) + b3 =================
    const float b3v = b3l[0];
    v8f a30 = (v8f){b3v, 0.f, 0.f, 0.f, 0.f, 0.f, 0.f, 0.f};  // M=0 row carries b3
    v8f a31 = a30;
#pragma unroll
    for (int kk = 0; kk < 4; ++kk) {
        Frag16 WA;  // w3 lives in A row M=0 only (lanes 0 and 16)
        if (row == 0) {
            const bf16* wp = w3l + kk * 32 + khalf * 16;
            WA.u[0] = *(const v4u*)wp;
            WA.u[1] = *(const v4u*)(wp + 8);
        } else {
            WA.u[0] = (v4u){0u, 0u, 0u, 0u};
            WA.u[1] = (v4u){0u, 0u, 0u, 0u};
        }
        Frag16 B0 = mkB(o20, kk, khalf);
        Frag16 B1 = mkB(o21, kk, khalf);
        a30 = wmma_bf16(WA, B0, a30);
        a31 = wmma_bf16(WA, B1, a31);
    }
    // D VGPR0, lanes 0..15 = (M=0, N=edge): one coalesced 64B store per tile
    if (lane < 16) {
        long long ea = edge0 + lane;
        long long eb = edge0 + 16 + lane;
        if (ea < nE) out[ea] = a30[0];
        if (eb < nE) out[eb] = a31[0];
    }
}

extern "C" void kernel_launch(void* const* d_in, const int* in_sizes, int n_in,
                              void* d_out, int out_size, void* d_ws, size_t ws_size,
                              hipStream_t stream) {
    (void)n_in; (void)d_ws; (void)ws_size; (void)out_size;
    const float*     x1 = (const float*)d_in[0];
    const float*     x2 = (const float*)d_in[1];
    const long long* ei = (const long long*)d_in[2];   // int64 edge_index [2][E]
    const float*     W1 = (const float*)d_in[3];
    const float*     b1 = (const float*)d_in[4];
    const float*     W2 = (const float*)d_in[5];
    const float*     b2 = (const float*)d_in[6];
    const float*     W3 = (const float*)d_in[7];
    const float*     b3 = (const float*)d_in[8];
    float* out = (float*)d_out;

    const long long nE = (long long)(in_sizes[2] / 2);
    const int blocks = (int)((nE + EDGES_PER_BLOCK - 1) / EDGES_PER_BLOCK);

    hipFuncSetAttribute((const void*)edge_mlp_wmma,
                        hipFuncAttributeMaxDynamicSharedMemorySize, LDS_TOTAL);
    edge_mlp_wmma<<<blocks, 256, LDS_TOTAL, stream>>>(
        x1, x2, ei, W1, b1, W2, b2, W3, b3, out, nE);
}
